// GCN_10591389352000
// MI455X (gfx1250) — matrix operand order repaired and checked by reading, not compiled, not run on hardware
//
#include <hip/hip_runtime.h>
#include <hip/hip_bf16.h>

typedef __attribute__((ext_vector_type(16))) __bf16 v16bf;
typedef __attribute__((ext_vector_type(8)))  float  v8f;

// ---------------------------------------------------------------------------
// Utility kernels
// ---------------------------------------------------------------------------
__global__ __launch_bounds__(256) void fill_f32(float* __restrict__ p, float v, size_t n) {
    size_t i = (size_t)blockIdx.x * 256 + threadIdx.x;
    if (i < n) p[i] = v;
}

__global__ __launch_bounds__(256) void deg_kernel(const int* __restrict__ dst,
                                                  float* __restrict__ deg, int E) {
    int e = blockIdx.x * 256 + threadIdx.x;
    if (e < E) atomicAdd(&deg[dst[e]], 1.0f);
}

__global__ __launch_bounds__(256) void rsqrt_kernel(float* __restrict__ p, int n) {
    int i = blockIdx.x * 256 + threadIdx.x;
    if (i < n) p[i] = rsqrtf(p[i]);
}

// ---------------------------------------------------------------------------
// WMMA GEMM:  Y[nrows x NCOL] = X[nrows x 128] @ W[128 x NCOL] (+ bias)
// block = 256 threads = 8 wave32 waves; each wave computes a 16-row strip.
//
// LDS layout for W (bf16): fragment-contiguous. A lane (col, half) in K-step
// kt needs K = kt*32 + half*16 + [0..15] for its column, so store
//   Bs[(col*8 + kchunk)*16 + (k & 15)],  kchunk = k >> 4   (8 chunks of 16 K)
// -> each B fragment is one aligned 32-byte chunk (2x ds_load_b128).
// ---------------------------------------------------------------------------
template <int NT, int NCOL, bool BIAS>
__global__ __launch_bounds__(256)
void gemm_wmma(const float* __restrict__ X, const float* __restrict__ W,
               const float* __restrict__ bias, float* __restrict__ Y, int nrows) {
    constexpr int NPAD = NT * 16;
    __shared__ __align__(32) __bf16 Bs[NPAD * 8 * 16];

    const int tid = threadIdx.x;
    // Stage W -> LDS (bf16, swizzled). i = kchunk*NPAD + col so consecutive
    // threads read consecutive columns (coalesced global loads).
    for (int i = tid; i < NPAD * 8; i += 256) {
        const int kc  = i / NPAD;        // kchunk 0..7
        const int col = i % NPAD;        // 0..NPAD-1
        __bf16 frag[16];
#pragma unroll
        for (int j = 0; j < 16; ++j) {
            const int k = kc * 16 + j;
            float w = (col < NCOL) ? W[k * NCOL + col] : 0.0f;
            frag[j] = (__bf16)w;
        }
        uint4* dst = reinterpret_cast<uint4*>(&Bs[(size_t)(col * 8 + kc) * 16]);
        const uint4* srcp = reinterpret_cast<const uint4*>(frag);
        dst[0] = srcp[0];
        dst[1] = srcp[1];
    }
    __syncthreads();

    const int wave = tid >> 5;
    const int lane = tid & 31;
    const int half = lane >> 4;   // 0: lanes 0-15, 1: lanes 16-31
    const int l16  = lane & 15;
    const int row0 = blockIdx.x * 128 + wave * 16;

    v8f acc[NT];
#pragma unroll
    for (int t = 0; t < NT; ++t) acc[t] = (v8f){0.f, 0.f, 0.f, 0.f, 0.f, 0.f, 0.f, 0.f};

    int rowA = row0 + l16;
    if (rowA >= nrows) rowA = nrows - 1;          // clamp; stores are guarded
    const float* xrow = X + (size_t)rowA * 128;

#pragma unroll
    for (int kt = 0; kt < 4; ++kt) {
        // A tile 16x32 bf16: elem e -> K = kbaseA+e
        const int kbaseA = kt * 32 + half * 16;
        const float4* pa = reinterpret_cast<const float4*>(xrow + kbaseA);
        float4 q0 = pa[0], q1 = pa[1];            // K = kbaseA .. kbaseA+7
        float4 q2 = pa[2], q3 = pa[3];            // K = kbaseA+8 .. kbaseA+15
        v16bf a;
        a[0] = (__bf16)q0.x; a[1] = (__bf16)q0.y; a[2]  = (__bf16)q0.z; a[3]  = (__bf16)q0.w;
        a[4] = (__bf16)q1.x; a[5] = (__bf16)q1.y; a[6]  = (__bf16)q1.z; a[7]  = (__bf16)q1.w;
        a[8] = (__bf16)q2.x; a[9] = (__bf16)q2.y; a[10] = (__bf16)q2.z; a[11] = (__bf16)q2.w;
        a[12] = (__bf16)q3.x; a[13] = (__bf16)q3.y; a[14] = (__bf16)q3.z; a[15] = (__bf16)q3.w;

        // B tile 32x16 bf16: elem e -> K = kt*32 + half*16 + e, col = l16
        const int kchunk = kt * 2 + half;
#pragma unroll
        for (int t = 0; t < NT; ++t) {
            const int col = t * 16 + l16;
            const v16bf b = *reinterpret_cast<const v16bf*>(
                &Bs[(size_t)(col * 8 + kchunk) * 16]);
            acc[t] = __builtin_amdgcn_wmma_f32_16x16x32_bf16(
                false, a, false, b, (short)0, acc[t], false, false);
        }
    }

    // C/D layout: VGPR r, lanes 0-15 -> M=r, lanes 16-31 -> M=r+8, N=l16
#pragma unroll
    for (int t = 0; t < NT; ++t) {
        const int col = t * 16 + l16;
        if (col < NCOL) {
            float badd = BIAS ? bias[col] : 0.0f;
#pragma unroll
            for (int r = 0; r < 8; ++r) {
                int orow = row0 + r + half * 8;
                if (orow < nrows) Y[(size_t)orow * NCOL + col] = acc[t][r] + badd;
            }
        }
    }
}

// ---------------------------------------------------------------------------
// Edge aggregation: one wave per edge; agg[dst] += h[src] * dinv[src]*dinv[dst]
// ---------------------------------------------------------------------------
__global__ __launch_bounds__(256)
void edge_agg(const float* __restrict__ h, const int* __restrict__ src,
              const int* __restrict__ dst, const float* __restrict__ dinv,
              float* __restrict__ agg, int E) {
    int e = blockIdx.x * 8 + (threadIdx.x >> 5);
    if (e >= E) return;
    int lane = threadIdx.x & 31;
    int s = src[e], d = dst[e];
    float c = dinv[s] * dinv[d];
    const float4* hp = reinterpret_cast<const float4*>(h + (size_t)s * 128);
    float4 v = hp[lane];
    float* ap = agg + (size_t)d * 128 + lane * 4;
    atomicAdd(ap + 0, v.x * c);
    atomicAdd(ap + 1, v.y * c);
    atomicAdd(ap + 2, v.z * c);
    atomicAdd(ap + 3, v.w * c);
}

// ---------------------------------------------------------------------------
// Fused: agg += h*dinv^2 + b (self-loop + bias), accumulate BN sums/sumsq
// block covers 32 rows x 128 channels with 256 threads (2 threads/channel)
// ---------------------------------------------------------------------------
__global__ __launch_bounds__(256)
void combine_stats(float* __restrict__ agg, const float* __restrict__ h,
                   const float* __restrict__ dinv, const float* __restrict__ b,
                   float* __restrict__ sums, float* __restrict__ sumsq, int nrows) {
    __shared__ float s1[256], s2[256];
    int tid = threadIdx.x;
    int ch = tid & 127;
    int part = tid >> 7;
    int rbase = blockIdx.x * 32;
    float bc = b[ch];
    float lsum = 0.f, lsq = 0.f;
    for (int r = rbase + part; r < rbase + 32; r += 2) {
        if (r < nrows) {
            size_t idx = (size_t)r * 128 + ch;
            float di = dinv[r];
            float v = agg[idx] + h[idx] * di * di + bc;
            agg[idx] = v;
            lsum += v; lsq += v * v;
        }
    }
    s1[tid] = lsum; s2[tid] = lsq;
    __syncthreads();
    if (tid < 128) {
        atomicAdd(&sums[ch],  s1[tid] + s1[tid + 128]);
        atomicAdd(&sumsq[ch], s2[tid] + s2[tid + 128]);
    }
}

__global__ void bn_finalize(const float* __restrict__ sums, const float* __restrict__ sumsq,
                            const float* __restrict__ gamma, const float* __restrict__ beta,
                            float* __restrict__ scale, float* __restrict__ shift, float invN) {
    int ch = threadIdx.x;
    float mean = sums[ch] * invN;
    float var  = sumsq[ch] * invN - mean * mean;
    float s = rsqrtf(var + 1e-5f) * gamma[ch];
    scale[ch] = s;
    shift[ch] = beta[ch] - mean * s;
}

__global__ __launch_bounds__(256)
void bn_relu(const float* __restrict__ in, float* __restrict__ out,
             const float* __restrict__ scale, const float* __restrict__ shift, size_t total) {
    size_t i = (size_t)blockIdx.x * 256 + threadIdx.x;
    if (i < total) {
        int ch = (int)(i & 127);
        float v = in[i] * scale[ch] + shift[ch];
        out[i] = v > 0.f ? v : 0.f;
    }
}

// ---------------------------------------------------------------------------
// Host-side orchestration
// ---------------------------------------------------------------------------
extern "C" void kernel_launch(void* const* d_in, const int* in_sizes, int n_in,
                              void* d_out, int out_size, void* d_ws, size_t ws_size,
                              hipStream_t stream) {
    const float* x   = (const float*)d_in[0];
    const int* esrc  = (const int*)d_in[1];
    const int* edst  = (const int*)d_in[2];
    const float* W1  = (const float*)d_in[3];
    const float* b1  = (const float*)d_in[4];
    const float* g1  = (const float*)d_in[5];
    const float* bt1 = (const float*)d_in[6];
    const float* W2  = (const float*)d_in[7];
    const float* b2  = (const float*)d_in[8];
    const float* g2  = (const float*)d_in[9];
    const float* bt2 = (const float*)d_in[10];
    const float* Wr  = (const float*)d_in[11];
    const float* br  = (const float*)d_in[12];
    float* out = (float*)d_out;

    const int N = in_sizes[0] / 128;
    const int E = in_sizes[1];
    const size_t NH = (size_t)N * 128;

    float* wf    = (float*)d_ws;
    float* dinv  = wf;                        // N floats (N multiple of 16)
    float* bufA  = dinv + N;                  // N*128
    float* bufB  = bufA + NH;                 // N*128
    float* stats = bufB + NH;                 // 512: sums, sumsq, scale, shift
    float* sums  = stats;
    float* sumsq = stats + 128;
    float* scale = stats + 256;
    float* shift = stats + 384;

    const int gN     = (N + 255) / 256;
    const int gNH    = (int)((NH + 255) / 256);
    const int gE     = (E + 255) / 256;
    const int gEdge  = (E + 7) / 8;
    const int gGemm  = (N + 127) / 128;
    const int gStat  = (N + 31) / 32;
    const float invN = 1.0f / (float)N;

    // degrees: start at 1.0 (self loop), add per incoming edge, rsqrt
    fill_f32<<<gN, 256, 0, stream>>>(dinv, 1.0f, (size_t)N);
    deg_kernel<<<gE, 256, 0, stream>>>(edst, dinv, E);
    rsqrt_kernel<<<gN, 256, 0, stream>>>(dinv, N);

    // ---------------- layer 1 ----------------
    gemm_wmma<8, 128, false><<<gGemm, 256, 0, stream>>>(x, W1, nullptr, bufA, N);
    fill_f32<<<gNH, 256, 0, stream>>>(bufB, 0.0f, NH);
    edge_agg<<<gEdge, 256, 0, stream>>>(bufA, esrc, edst, dinv, bufB, E);
    fill_f32<<<1, 256, 0, stream>>>(stats, 0.0f, (size_t)256);
    combine_stats<<<gStat, 256, 0, stream>>>(bufB, bufA, dinv, b1, sums, sumsq, N);
    bn_finalize<<<1, 128, 0, stream>>>(sums, sumsq, g1, bt1, scale, shift, invN);
    bn_relu<<<gNH, 256, 0, stream>>>(bufB, bufA, scale, shift, NH);   // bufA = h1

    // ---------------- layer 2 ----------------
    gemm_wmma<8, 128, false><<<gGemm, 256, 0, stream>>>(bufA, W2, nullptr, bufB, N); // bufB = h1@W2
    fill_f32<<<gNH, 256, 0, stream>>>(bufA, 0.0f, NH);
    edge_agg<<<gEdge, 256, 0, stream>>>(bufB, esrc, edst, dinv, bufA, E);
    fill_f32<<<1, 256, 0, stream>>>(stats, 0.0f, (size_t)256);
    combine_stats<<<gStat, 256, 0, stream>>>(bufA, bufB, dinv, b2, sums, sumsq, N);
    bn_finalize<<<1, 128, 0, stream>>>(sums, sumsq, g2, bt2, scale, shift, invN);
    bn_relu<<<gNH, 256, 0, stream>>>(bufA, bufB, scale, shift, NH);   // bufB = h2

    // ---------------- readout ----------------
    gemm_wmma<3, 40, true><<<gGemm, 256, 0, stream>>>(bufB, Wr, br, out, N);
}
